// GPT_15968688407122
// MI455X (gfx1250) — matrix operand order repaired and checked by reading, not compiled, not run on hardware
//
#include <hip/hip_runtime.h>
#include <hip/hip_bf16.h>
#include <math.h>

typedef __attribute__((ext_vector_type(16))) _Float16     v16h;
typedef __attribute__((ext_vector_type(8)))  float        v8f;
typedef __attribute__((ext_vector_type(4)))  unsigned int v4u;
typedef __attribute__((ext_vector_type(8)))  unsigned int v8u;

#define BB  32
#define CCH 512
#define LL  8
#define NHH 8
#define TT  784
#define DD  64
#define MM  (BB * TT)   // 25088 rows, divisible by 128
#define TP  800         // padded key length for Vt (multiple of 32)

// ---------------------------------------------------------------------------
// DPP16 16-lane-group reductions (stay within lanes 0-15 / 16-31 halves).
// ---------------------------------------------------------------------------
template <int CTRL>
static __device__ __forceinline__ float dppmov(float x) {
  return __int_as_float(__builtin_amdgcn_update_dpp(
      0, __float_as_int(x), CTRL, 0xf, 0xf, true));
}
static __device__ __forceinline__ float red16_max(float x) {
  x = fmaxf(x, dppmov<0xB1>(x));
  x = fmaxf(x, dppmov<0x4E>(x));
  x = fmaxf(x, dppmov<0x141>(x));
  x = fmaxf(x, dppmov<0x140>(x));
  return x;
}
static __device__ __forceinline__ float red16_sum(float x) {
  x += dppmov<0xB1>(x);
  x += dppmov<0x4E>(x);
  x += dppmov<0x141>(x);
  x += dppmov<0x140>(x);
  return x;
}

// ---------------------------------------------------------------------------
// WMMA fragment loaders (contiguous k = 16*(lane/16) + s, not AMD's table order; wave32)
// ---------------------------------------------------------------------------
static __device__ __forceinline__ v16h load_a_frag(const _Float16* p0, int ld) {
  int lane = threadIdx.x & 31;
  const _Float16* p = p0 + (size_t)(lane & 15) * ld + ((lane & 16) ? 16 : 0);
  union { v16h v; float4 f[2]; } u;
  u.f[0] = *reinterpret_cast<const float4*>(p);
  u.f[1] = *reinterpret_cast<const float4*>(p + 8);
  return u.v;
}
static __device__ __forceinline__ v16h load_b_frag(const _Float16* p0, int ld) {
  int lane = threadIdx.x & 31;
  const _Float16* p = p0 + (size_t)(lane & 15) * ld + ((lane & 16) ? 16 : 0);
  union { v16h v; float4 f[2]; } u;
  u.f[0] = *reinterpret_cast<const float4*>(p);
  u.f[1] = *reinterpret_cast<const float4*>(p + 8);
  return u.v;
}

// ---------------------------------------------------------------------------
// Tensor Data Mover: 2D tile (tile_h rows x tile_w elems of 2 bytes) from
// global (row stride = row_stride_elems) into contiguous LDS.  D# per
// CDNA5 ISA: group0 = {count, lds_addr, global_addr,
// type=2}; group1 = {data_size=1, tile dims, dim0 stride; huge tensor dims
// so no OOB clipping; no multicast/pad/iterate}.  Issued once per wave;
// tracked by TENSORcnt.
// ---------------------------------------------------------------------------
static __device__ __forceinline__ void tdm_load_2d(
    unsigned lds_byte_addr, const void* gptr, unsigned tile_w_elems,
    unsigned tile_h, unsigned row_stride_elems) {
  unsigned long long ga = (unsigned long long)(size_t)gptr;
  v4u g0;
  g0[0] = 1u;                                    // count=1 (valid user D#)
  g0[1] = lds_byte_addr;                         // lds_addr  (bits 63:32)
  g0[2] = (unsigned)ga;                          // global_addr[31:0]
  g0[3] = ((unsigned)(ga >> 32) & 0x01FFFFFFu)   // global_addr[56:32]
          | 0x80000000u;                         // type=2 ("image") bits 127:126
  v8u g1;
  g1[0] = 0x00010000u;                           // data_size=1 (2B); mask=0
  g1[1] = 0xFFFF0000u;                           // tensor_dim0 = 0x7FFFFFFF lo16
  g1[2] = 0xFFFF7FFFu;                           // dim0 hi16 | tensor_dim1 lo16
  g1[3] = 0x00007FFFu | (tile_w_elems << 16);    // dim1 hi16 | tile_dim0
  g1[4] = tile_h;                                // tile_dim1 ; tile_dim2=0
  g1[5] = row_stride_elems;                      // tensor_dim0_stride lo32
  g1[6] = 0u;                                    // stride hi16 | dim1_stride lo16
  g1[7] = 0u;                                    // dim1_stride hi32
  asm volatile("tensor_load_to_lds %0, %1" :: "s"(g0), "s"(g1) : "memory");
}

// ---------------------------------------------------------------------------
// Weight convert + transpose: dst[n*K+k] = (f16) src[k*N+n]
// ---------------------------------------------------------------------------
__global__ void convert_wT(_Float16* __restrict__ dst, const float* __restrict__ src,
                           int K, int N) {
  int idx = blockIdx.x * blockDim.x + threadIdx.x;
  if (idx >= N * K) return;
  int k = idx % K;
  int n = idx / K;
  dst[idx] = (_Float16)src[(size_t)k * N + n];
}

__global__ void concat3(float* __restrict__ dst, const float* __restrict__ a,
                        const float* __restrict__ b, const float* __restrict__ c, int n) {
  int i = blockIdx.x * blockDim.x + threadIdx.x;
  if (i < n) dst[i] = a[i];
  else if (i < 2 * n) dst[i] = b[i - n];
  else if (i < 3 * n) dst[i] = c[i - 2 * n];
}

// ---------------------------------------------------------------------------
// Embedding: 1x1 conv over [x, X(i), Y(j)] channels, then shift-right by one.
// ---------------------------------------------------------------------------
__global__ void embed_kernel(float* __restrict__ h, const float* __restrict__ x,
                             const float* __restrict__ w, const float* __restrict__ bias) {
  int idx = blockIdx.x * blockDim.x + threadIdx.x;   // over MM*CCH
  if (idx >= MM * CCH) return;
  int c = idx % CCH;
  int m = idx / CCH;
  int b = m / TT, t = m % TT;
  float v = 0.0f;
  if (t != 0) {
    int p = t - 1;
    int i = p / 28, j = p % 28;
    float xv = x[(size_t)b * TT + p];
    float xi = (float)i * (1.0f / 27.0f);
    float yj = (float)j * (1.0f / 27.0f);
    v = xv * w[0 * CCH + c] + xi * w[1 * CCH + c] + yj * w[2 * CCH + c] + bias[c];
  }
  h[idx] = v;
}

// ---------------------------------------------------------------------------
// LayerNorm(row of 512) -> f16
// ---------------------------------------------------------------------------
__global__ __launch_bounds__(256) void ln_kernel(_Float16* __restrict__ out,
    const float* __restrict__ hsrc, const float* __restrict__ g,
    const float* __restrict__ beta) {
  __shared__ float sbuf[256];
  int m = blockIdx.x;
  int t = threadIdx.x;
  const float* row = hsrc + (size_t)m * CCH;
  float x0 = row[t], x1 = row[t + 256];
  sbuf[t] = x0 + x1;
  __syncthreads();
  for (int st = 128; st > 0; st >>= 1) { if (t < st) sbuf[t] += sbuf[t + st]; __syncthreads(); }
  float mean = sbuf[0] * (1.0f / CCH);
  __syncthreads();
  float d0 = x0 - mean, d1 = x1 - mean;
  sbuf[t] = d0 * d0 + d1 * d1;
  __syncthreads();
  for (int st = 128; st > 0; st >>= 1) { if (t < st) sbuf[t] += sbuf[t + st]; __syncthreads(); }
  float rstd = rsqrtf(sbuf[0] * (1.0f / CCH) + 1e-5f);
  out[(size_t)m * CCH + t]       = (_Float16)(d0 * rstd * g[t] + beta[t]);
  out[(size_t)m * CCH + t + 256] = (_Float16)(d1 * rstd * g[t + 256] + beta[t + 256]);
}

// ---------------------------------------------------------------------------
// Final LN + head dot-product -> logits f32
// ---------------------------------------------------------------------------
__global__ __launch_bounds__(256) void head_kernel(float* __restrict__ out,
    const float* __restrict__ hsrc, const float* __restrict__ g,
    const float* __restrict__ beta, const float* __restrict__ hw) {
  __shared__ float sbuf[256];
  int m = blockIdx.x;
  int t = threadIdx.x;
  const float* row = hsrc + (size_t)m * CCH;
  float x0 = row[t], x1 = row[t + 256];
  sbuf[t] = x0 + x1;
  __syncthreads();
  for (int st = 128; st > 0; st >>= 1) { if (t < st) sbuf[t] += sbuf[t + st]; __syncthreads(); }
  float mean = sbuf[0] * (1.0f / CCH);
  __syncthreads();
  float d0 = x0 - mean, d1 = x1 - mean;
  sbuf[t] = d0 * d0 + d1 * d1;
  __syncthreads();
  for (int st = 128; st > 0; st >>= 1) { if (t < st) sbuf[t] += sbuf[t + st]; __syncthreads(); }
  float rstd = rsqrtf(sbuf[0] * (1.0f / CCH) + 1e-5f);
  __syncthreads();
  float v0 = (d0 * rstd * g[t] + beta[t]) * hw[t];
  float v1 = (d1 * rstd * g[t + 256] + beta[t + 256]) * hw[t + 256];
  sbuf[t] = v0 + v1;
  __syncthreads();
  for (int st = 128; st > 0; st >>= 1) { if (t < st) sbuf[t] += sbuf[t + st]; __syncthreads(); }
  if (t == 0) out[m] = sbuf[0];
}

// ---------------------------------------------------------------------------
// WMMA GEMM: out[M,N] = A[M,K](f16,row-major) x Bt[N,K](f16, pre-transposed) + bias
// MODE 0: out f16 ; MODE 1: out f16 with exact GELU ; MODE 2: f32 out += (residual)
// Block = 256 threads = 8 waves; tile 128x128, K-step 32.
// Staging: A panel via per-lane GLOBAL_LOAD_ASYNC_TO_LDS_B128 (ASYNCcnt);
// B panel via one TDM descriptor issued by wave 0 (TENSORcnt).
// Double-buffered: tile k+1 DMA overlaps tile k WMMAs; one barrier per K-step.
// ---------------------------------------------------------------------------
template <int MODE>
__global__ __launch_bounds__(256) void gemm_wmma(
    const _Float16* __restrict__ A, const _Float16* __restrict__ Bt,
    const float* __restrict__ bias, void* __restrict__ outv,
    int M, int N, int K) {
  __shared__ _Float16 sA[2][128 * 32];
  __shared__ _Float16 sB[2][128 * 32];
  const int tid  = threadIdx.x;
  const int wid  = tid >> 5;
  const int lane = tid & 31;
  const int m0 = blockIdx.y * 128;
  const int n0 = blockIdx.x * 128;

  v8f acc[8];
#pragma unroll
  for (int j = 0; j < 8; ++j) acc[j] = {};

  const int lrow   = tid >> 1;          // 0..127
  const int lchunk = (tid & 1) * 16;    // halves
  const unsigned ldsA0  = (unsigned)(size_t)(&sA[0][lrow * 32 + lchunk]);
  const unsigned ldsA1  = (unsigned)(size_t)(&sA[1][lrow * 32 + lchunk]);
  const unsigned ldsBb0 = (unsigned)(size_t)(&sB[0][0]);
  const unsigned ldsBb1 = (unsigned)(size_t)(&sB[1][0]);
  const _Float16* gArow  = A  + (size_t)(m0 + lrow) * K + lchunk;
  const _Float16* gBtile = Bt + (size_t)n0 * K;

  auto issue = [&](int buf, int k0) {
    const _Float16* gA = gArow + k0;
    unsigned la = buf ? ldsA1 : ldsA0;
    asm volatile("global_load_async_to_lds_b128 %0, %1, off"
                 :: "v"(la), "v"(gA) : "memory");
    asm volatile("global_load_async_to_lds_b128 %0, %1, off offset:16"
                 :: "v"(la), "v"(gA) : "memory");
    if (wid == 0)   // one TDM descriptor moves the whole 128x32 B panel
      tdm_load_2d(buf ? ldsBb1 : ldsBb0, gBtile + k0, 32u, 128u, (unsigned)K);
  };

  const int NT = K / 32;
  issue(0, 0);
  for (int kt = 0; kt < NT; ++kt) {
    const int cur = kt & 1;
    asm volatile("s_wait_asynccnt 0x0" ::: "memory");     // my A strip landed
    if (wid == 0) __builtin_amdgcn_s_wait_tensorcnt(0);   // B panel landed
    __syncthreads();   // tile visible everywhere; prior reads of other buf done
    if (kt + 1 < NT) issue(cur ^ 1, (kt + 1) * 32);       // overlap next DMA
    v16h af = load_a_frag(&sA[cur][(wid * 16) * 32], 32);
#pragma unroll
    for (int j = 0; j < 8; ++j) {
      v16h bf = load_b_frag(&sB[cur][(j * 16) * 32], 32);
      acc[j] = __builtin_amdgcn_wmma_f32_16x16x32_f16(false, af, false, bf,
                                                      (short)0, acc[j], false, false);
    }
  }

  const int mrow = m0 + wid * 16 + ((lane & 16) ? 8 : 0);
  const int ncol = n0 + (lane & 15);
#pragma unroll
  for (int j = 0; j < 8; ++j) {
    int n = ncol + j * 16;
    float bval = bias[n];
#pragma unroll
    for (int i = 0; i < 8; ++i) {
      int m = mrow + i;
      float v = acc[j][i] + bval;
      if (MODE == 0) {
        ((_Float16*)outv)[(size_t)m * N + n] = (_Float16)v;
      } else if (MODE == 1) {
        float ge = 0.5f * v * (1.0f + erff(v * 0.70710678118f));
        ((_Float16*)outv)[(size_t)m * N + n] = (_Float16)ge;
      } else {
        ((float*)outv)[(size_t)m * N + n] += v;
      }
    }
  }
}

// ---------------------------------------------------------------------------
// V transpose (per b,h): Vt[bh*64+d][s] = qkv[b*T+s][2C + h*64 + d], zero pad to TP
// ---------------------------------------------------------------------------
__global__ void transpose_v(_Float16* __restrict__ vt, const _Float16* __restrict__ qkv) {
  int idx = blockIdx.x * blockDim.x + threadIdx.x;   // BB*NHH*DD*TP
  if (idx >= BB * NHH * DD * TP) return;
  int s  = idx % TP;
  int r  = idx / TP;      // bh*DD + d
  int d  = r % DD;
  int bh = r / DD;
  int b = bh / NHH, h = bh % NHH;
  _Float16 v = (_Float16)0.0f;
  if (s < TT)
    v = qkv[((size_t)(b * TT + s)) * (3 * CCH) + 2 * CCH + h * DD + d];
  vt[idx] = v;
}

// ---------------------------------------------------------------------------
// Flash-attention: one wave per 16-query tile of one (b,h).
// ---------------------------------------------------------------------------
__global__ __launch_bounds__(256) void attn_kernel(
    const _Float16* __restrict__ qkv, const _Float16* __restrict__ vt,
    _Float16* __restrict__ y) {
  __shared__ _Float16 sP[8][16 * 32];
  const int wid  = threadIdx.x >> 5;
  const int lane = threadIdx.x & 31;
  const int bh = blockIdx.x;
  const int b = bh / NHH, h = bh % NHH;
  const int mt = blockIdx.y * 8 + wid;
  if (mt >= TT / 16) return;            // whole-wave exit: EXEC stays full for WMMA
  const int m0 = mt * 16;
  const int ld = 3 * CCH;

  const _Float16* qbase = qkv + ((size_t)(b * TT + m0)) * ld + h * DD;
  v16h qa0 = load_a_frag(qbase, ld);        // d = 0..31
  v16h qa1 = load_a_frag(qbase + 32, ld);   // d = 32..63
  const _Float16* kbase = qkv + (size_t)b * TT * ld + CCH + h * DD;
  const _Float16* vbase = vt + (size_t)(bh * DD) * TP;

  v8f O[4];
#pragma unroll
  for (int j = 0; j < 4; ++j) O[j] = {};
  float rm[8], rs[8];
#pragma unroll
  for (int i = 0; i < 8; ++i) { rm[i] = -1e30f; rs[i] = 0.0f; }

  const int col   = lane & 15;
  const int rbase = (lane & 16) ? 8 : 0;
  _Float16* pbuf = sP[wid];

  for (int s0 = 0; s0 <= mt; s0 += 2) {
    v8f S[2];
#pragma unroll
    for (int tsel = 0; tsel < 2; ++tsel) {
      int st = s0 + tsel;
      int srow0 = st * 16;
      int krow = srow0 + col;
      if (krow > TT - 1) krow = TT - 1;    // clamp (masked out below anyway)
      const _Float16* kp = kbase + (size_t)krow * ld + ((lane & 16) ? 16 : 0);
      union { v16h v; float4 f[2]; } kb0, kb1;
      kb0.f[0] = *(const float4*)(kp);
      kb0.f[1] = *(const float4*)(kp + 8);
      kb1.f[0] = *(const float4*)(kp + 32);
      kb1.f[1] = *(const float4*)(kp + 40);
      v8f s = {};
      s = __builtin_amdgcn_wmma_f32_16x16x32_f16(false, qa0, false, kb0.v, (short)0, s, false, false);
      s = __builtin_amdgcn_wmma_f32_16x16x32_f16(false, qa1, false, kb1.v, (short)0, s, false, false);
#pragma unroll
      for (int i = 0; i < 8; ++i) {
        int mrow = m0 + rbase + i;
        int scol = srow0 + col;
        float val = s[i] * 0.125f;         // 1/sqrt(64)
        s[i] = (scol <= mrow) ? val : -1e30f;
      }
      S[tsel] = s;
    }

    // online softmax update: row stats via DPP16 reductions (VALU only)
#pragma unroll
    for (int i = 0; i < 8; ++i) {
      float t = red16_max(fmaxf(S[0][i], S[1][i]));
      float nm = fmaxf(rm[i], t);
      float corr = __expf(rm[i] - nm);
      rm[i] = nm;
      rs[i] *= corr;
#pragma unroll
      for (int j = 0; j < 4; ++j) O[j][i] *= corr;
      float p0 = __expf(S[0][i] - nm);
      float p1 = __expf(S[1][i] - nm);
      rs[i] += red16_sum(p0 + p1);
      int row = rbase + i;
      pbuf[row * 32 + col]      = (_Float16)p0;
      pbuf[row * 32 + 16 + col] = (_Float16)p1;
    }

    // P (16x32) as A-fragment from LDS; Vt rows contiguous in s -> B-fragments
    v16h pa = load_a_frag(pbuf, 32);
#pragma unroll
    for (int j = 0; j < 4; ++j) {
      const _Float16* vp = vbase + (size_t)(j * 16) * TP + s0 * 16;
      v16h vb = load_b_frag(vp, TP);
      O[j] = __builtin_amdgcn_wmma_f32_16x16x32_f16(false, pa, false, vb, (short)0, O[j], false, false);
    }
  }

#pragma unroll
  for (int i = 0; i < 8; ++i) {
    float inv = 1.0f / rs[i];
    int m = b * TT + m0 + rbase + i;
#pragma unroll
    for (int j = 0; j < 4; ++j)
      y[(size_t)m * CCH + h * DD + j * 16 + col] = (_Float16)(O[j][i] * inv);
  }
}

// ---------------------------------------------------------------------------
// Host orchestration
// ---------------------------------------------------------------------------
extern "C" void kernel_launch(void* const* d_in, const int* in_sizes, int n_in,
                              void* d_out, int out_size, void* d_ws, size_t ws_size,
                              hipStream_t stream) {
  (void)in_sizes; (void)n_in; (void)out_size; (void)ws_size;
  const float* x      = (const float*)d_in[0];
  const float* emb_w  = (const float*)d_in[1];
  const float* emb_b  = (const float*)d_in[2];
  const float* ln1_g  = (const float*)d_in[3];
  const float* ln1_b  = (const float*)d_in[4];
  const float* Wq     = (const float*)d_in[5];
  const float* bq     = (const float*)d_in[6];
  const float* Wk     = (const float*)d_in[7];
  const float* bk     = (const float*)d_in[8];
  const float* Wv     = (const float*)d_in[9];
  const float* bv     = (const float*)d_in[10];
  const float* Wp     = (const float*)d_in[11];
  const float* bp     = (const float*)d_in[12];
  const float* ln2_g  = (const float*)d_in[13];
  const float* ln2_b  = (const float*)d_in[14];
  const float* W1     = (const float*)d_in[15];
  const float* b1     = (const float*)d_in[16];
  const float* W2     = (const float*)d_in[17];
  const float* b2     = (const float*)d_in[18];
  const float* lnf_g  = (const float*)d_in[19];
  const float* lnf_b  = (const float*)d_in[20];
  const float* head_w = (const float*)d_in[21];

  char* ws = (char*)d_ws;
  size_t off = 0;
  auto take = [&](size_t bytes) -> void* {
    void* p = ws + off;
    off += (bytes + 255) & ~(size_t)255;
    return p;
  };
  float*    hbuf  = (float*)take((size_t)MM * CCH * 4);
  _Float16* a16   = (_Float16*)take((size_t)MM * CCH * 2);
  _Float16* qkv16 = (_Float16*)take((size_t)MM * 3 * CCH * 2);
  _Float16* vt16  = (_Float16*)take((size_t)BB * NHH * DD * TP * 2);
  _Float16* y16   = (_Float16*)take((size_t)MM * CCH * 2);
  _Float16* wqkvt = (_Float16*)take((size_t)LL * 3 * CCH * CCH * 2);
  float*    bqkv  = (float*)take((size_t)LL * 3 * CCH * 4);
  _Float16* wpt   = (_Float16*)take((size_t)LL * CCH * CCH * 2);
  _Float16* w1t   = (_Float16*)take((size_t)LL * CCH * 4 * CCH * 2);
  _Float16* w2t   = (_Float16*)take((size_t)LL * 4 * CCH * CCH * 2);
  // MLP hidden aliases the (dead at that point) qkv/vt/y region
  _Float16* u16   = qkv16;

  const int CB = 256;
  const size_t wsz  = (size_t)CCH * CCH;        // 262144
  const size_t wsz4 = (size_t)CCH * 4 * CCH;    // 1048576

  // --- per-call weight convert + transpose ---
  for (int l = 0; l < LL; ++l) {
    _Float16* wq3 = wqkvt + (size_t)l * 3 * wsz;
    convert_wT<<<(int)((wsz  + CB - 1) / CB), CB, 0, stream>>>(wq3,            Wq + l * wsz, CCH, CCH);
    convert_wT<<<(int)((wsz  + CB - 1) / CB), CB, 0, stream>>>(wq3 + wsz,      Wk + l * wsz, CCH, CCH);
    convert_wT<<<(int)((wsz  + CB - 1) / CB), CB, 0, stream>>>(wq3 + 2 * wsz,  Wv + l * wsz, CCH, CCH);
    concat3<<<(3 * CCH + CB - 1) / CB, CB, 0, stream>>>(bqkv + l * 3 * CCH,
                                                        bq + l * CCH, bk + l * CCH, bv + l * CCH, CCH);
    convert_wT<<<(int)((wsz  + CB - 1) / CB), CB, 0, stream>>>(wpt + l * wsz,  Wp + l * wsz, CCH, CCH);
    convert_wT<<<(int)((wsz4 + CB - 1) / CB), CB, 0, stream>>>(w1t + l * wsz4, W1 + l * wsz4, CCH, 4 * CCH);
    convert_wT<<<(int)((wsz4 + CB - 1) / CB), CB, 0, stream>>>(w2t + l * wsz4, W2 + l * wsz4, 4 * CCH, CCH);
  }

  // --- embedding + shift ---
  embed_kernel<<<(MM * CCH + CB - 1) / CB, CB, 0, stream>>>(hbuf, x, emb_w, emb_b);

  // --- transformer layers ---
  for (int l = 0; l < LL; ++l) {
    ln_kernel<<<MM, 256, 0, stream>>>(a16, hbuf, ln1_g + l * CCH, ln1_b + l * CCH);

    gemm_wmma<0><<<dim3(12, MM / 128), 256, 0, stream>>>(
        a16, wqkvt + (size_t)l * 3 * wsz, bqkv + l * 3 * CCH, qkv16, MM, 3 * CCH, CCH);

    transpose_v<<<(BB * NHH * DD * TP) / CB, CB, 0, stream>>>(vt16, qkv16);

    attn_kernel<<<dim3(BB * NHH, 7), 256, 0, stream>>>(qkv16, vt16, y16);

    gemm_wmma<2><<<dim3(4, MM / 128), 256, 0, stream>>>(
        y16, wpt + (size_t)l * wsz, bp + l * CCH, hbuf, MM, CCH, CCH);

    ln_kernel<<<MM, 256, 0, stream>>>(a16, hbuf, ln2_g + l * CCH, ln2_b + l * CCH);

    gemm_wmma<1><<<dim3(16, MM / 128), 256, 0, stream>>>(
        a16, w1t + (size_t)l * wsz4, b1 + (size_t)l * 4 * CCH, u16, MM, 4 * CCH, CCH);

    gemm_wmma<2><<<dim3(4, MM / 128), 256, 0, stream>>>(
        u16, w2t + (size_t)l * wsz4, b2 + l * CCH, hbuf, MM, CCH, 4 * CCH);
  }

  // --- final LN + head ---
  head_kernel<<<MM, 256, 0, stream>>>((float*)d_out, hbuf, lnf_g, lnf_b, head_w);
}
